// GNNRiskModel_17540646437368
// MI455X (gfx1250) — matrix operand order repaired and checked by reading, not compiled, not run on hardware
//
#include <hip/hip_runtime.h>
#include <hip/hip_bf16.h>
#include <math.h>

// ---------------------------------------------------------------------------
// Types for CDNA5 WMMA
// ---------------------------------------------------------------------------
typedef __attribute__((ext_vector_type(16))) __bf16 v16bf;
typedef __attribute__((ext_vector_type(8)))  float  v8f;

union BfVec {
    uint4 q[2];
    v16bf v;
};

#define H_DIM   128
#define F_IN    32
#define HEADS   4
#define HD      32

// CDNA5 async Global->LDS copy path (ASYNCcnt-tracked), with portable fallback.
#if defined(__AMDGCN__) && __has_builtin(__builtin_amdgcn_global_load_async_to_lds_b128)
#define ASYNC_COPY 1
// Matches the builtin's parameter type: int vector of 4, addrspace-qualified.
typedef int v4i_vec __attribute__((vector_size(16)));
typedef __attribute__((address_space(1))) v4i_vec* gptr_b128;
typedef __attribute__((address_space(3))) v4i_vec* lptr_b128;
#else
#define ASYNC_COPY 0
#endif

__device__ __forceinline__ float leaky02(float x) { return x > 0.0f ? x : 0.2f * x; }

__device__ __forceinline__ void atomicMaxFloat(float* addr, float val) {
    // Sign-aware float max via integer atomics (works for mixed signs).
    if (val >= 0.0f)
        atomicMax((int*)addr, __float_as_int(val));
    else
        atomicMin((unsigned int*)addr, __float_as_uint(val));
}

// ---------------------------------------------------------------------------
// Elementwise / setup kernels
// ---------------------------------------------------------------------------
__global__ void k_zero_f32(float* __restrict__ p, int n) {
    int i = blockIdx.x * blockDim.x + threadIdx.x;
    if (i < n) p[i] = 0.0f;
}

__global__ void k_count_deg(const int* __restrict__ dst, float* __restrict__ deg, int nE) {
    int i = blockIdx.x * blockDim.x + threadIdx.x;
    if (i < nE) atomicAdd(&deg[dst[i]], 1.0f);
}

__global__ void k_finalize_dinv(float* __restrict__ deg, int n) {
    int i = blockIdx.x * blockDim.x + threadIdx.x;
    if (i < n) deg[i] = rsqrtf(deg[i] + 1.0f);   // +1 for self-loop
}

__global__ void k_cvt_f32_bf16(const float* __restrict__ in, __bf16* __restrict__ out, int n) {
    int i = blockIdx.x * blockDim.x + threadIdx.x;
    if (i < n) out[i] = (__bf16)in[i];
}

// Pack f32 weight matrix [K, Ncols] into this file's per-lane WMMA B operand
// layout (contiguous k, not the ISA table order; 16-bit B 32x16): lanes 0-15 hold K=0..15 (2/VGPR),
// lanes 16-31 hold K=16..31.  Output index = ((chunk*T + tile)*32 + lane)*8 + j.
__global__ void k_pack_w_bf16(const float* __restrict__ W, unsigned int* __restrict__ Wp,
                              int K, int Ncols) {
    int idx = blockIdx.x * blockDim.x + threadIdx.x;
    int T = Ncols >> 4;
    int total = (K >> 5) * T * 32 * 8;
    if (idx >= total) return;
    int j    = idx & 7;
    int lane = (idx >> 3) & 31;
    int rest = idx >> 8;
    int t    = rest % T;
    int c    = rest / T;
    int k0 = (c << 5) + ((lane >> 4) << 4) + (j << 1);
    int n  = (t << 4) + (lane & 15);
    union { __bf16 h[2]; unsigned int u; } pk;
    pk.h[0] = (__bf16)W[(size_t)k0 * Ncols + n];
    pk.h[1] = (__bf16)W[(size_t)(k0 + 1) * Ncols + n];
    Wp[idx] = pk.u;
}

// ---------------------------------------------------------------------------
// bf16 WMMA GEMM:  Out[M,Ncols](f32) = A[M,K](bf16, row-major) * Wp (pre-packed)
// Each wave: 16 rows x 64 cols (4 accumulator tiles). Block = 4 waves.
// The block's packed-B working set (<=16KB) is staged in LDS once (async
// global->LDS when available), then the K-loop runs on ds_load_b128 only.
// ---------------------------------------------------------------------------
__global__ __launch_bounds__(128)
void k_wmma_gemm_bf16(const __bf16* __restrict__ A, const unsigned int* __restrict__ Bp,
                      float* __restrict__ Out, int M, int K, int Ncols) {
    __shared__ __align__(16) unsigned int sB[4 * 4 * 32 * 8];   // up to K=128 -> 16KB

    const int tid     = threadIdx.x;
    const int T       = Ncols >> 4;
    const int nchunks = K >> 5;
    const int tn0     = blockIdx.y << 2;          // first of 4 N-tiles for this block

    // --- cooperative B staging: LDS uint4 index q == ((c*4+t)*32+lane)*2 + j4
    const int nq = nchunks << 8;                  // uint4 count (256 per chunk)
    for (int q = tid; q < nq; q += 128) {
        const int j4   = q & 1;
        const int lane = (q >> 1) & 31;
        const int t    = (q >> 6) & 3;
        const int c    = q >> 8;
        const size_t g = ((size_t)(c * T + tn0 + t) * 32 + lane) * 2 + j4;
#if ASYNC_COPY
        __builtin_amdgcn_global_load_async_to_lds_b128(
            (gptr_b128)((const uint4*)Bp + g),
            (lptr_b128)((uint4*)sB + q),
            0, 0);
#else
        ((uint4*)sB)[q] = ((const uint4*)Bp)[g];
#endif
    }
#if ASYNC_COPY
#if __has_builtin(__builtin_amdgcn_s_wait_asynccnt)
    __builtin_amdgcn_s_wait_asynccnt(0);
#else
    asm volatile("s_wait_asynccnt 0x0" ::: "memory");
#endif
#endif
    __syncthreads();

    // --- compute phase (guarded; barrier above is reached by all threads)
    const int wave   = tid >> 5;
    const int lane   = tid & 31;
    const int tile_m = (blockIdx.x * 4 + wave) << 4;
    if (tile_m < M) {
        const int half = lane >> 4;
        const size_t rowoff = (size_t)(tile_m + (lane & 15)) * K;

        v8f acc[4] = {};
        for (int c = 0; c < nchunks; ++c) {
            const int kb = c << 5;
            // A operand: contiguous k = 16*half + s (matches packed B) -> two 16B loads/lane
            BfVec a;
            a.q[0] = *(const uint4*)(A + rowoff + kb + 16 * half);
            a.q[1] = *(const uint4*)(A + rowoff + kb + 16 * half + 8);
            const uint4* bbase = (const uint4*)sB + ((size_t)(c * 4) * 32 + lane) * 2;
#pragma unroll
            for (int t = 0; t < 4; ++t) {
                BfVec b;
                b.q[0] = bbase[t * 64 + 0];       // ds_load_b128, immediate offsets
                b.q[1] = bbase[t * 64 + 1];
                acc[t] = __builtin_amdgcn_wmma_f32_16x16x32_bf16(
                    false, a.v, false, b.v, (short)0, acc[t], false, false);
            }
        }
        // C/D layout: lane -> col (lane&15), VGPR j -> row j + 8*(lane>>4)
#pragma unroll
        for (int t = 0; t < 4; ++t) {
            const int col = ((tn0 + t) << 4) + (lane & 15);
#pragma unroll
            for (int j = 0; j < 8; ++j) {
                const int row = tile_m + (half << 3) + j;
                Out[(size_t)row * Ncols + col] = acc[t][j];
            }
        }
    }
}

// ---------------------------------------------------------------------------
// GCN aggregation
// ---------------------------------------------------------------------------
__global__ void k_gcn_self_init(const float* __restrict__ hlin, const float* __restrict__ dinv,
                                float* __restrict__ out, int total) {
    int i = blockIdx.x * blockDim.x + threadIdx.x;
    if (i >= total) return;
    int n = i >> 7;
    float di = dinv[n];
    out[i] = hlin[i] * di * di;
}

__global__ void k_gcn_edge_agg(const float* __restrict__ h, const int* __restrict__ src,
                               const int* __restrict__ dst, const float* __restrict__ dinv,
                               float* __restrict__ out, int nE) {
    int t = blockIdx.x * blockDim.x + threadIdx.x;
    int e = t >> 5;
    if (e >= nE) return;
    int lane = t & 31;
    int s = src[e], d = dst[e];
    float nrm = dinv[s] * dinv[d];
    float4 v = *(const float4*)(h + (size_t)s * H_DIM + lane * 4);
    float* o = out + (size_t)d * H_DIM + lane * 4;
    atomicAdd(o + 0, v.x * nrm);
    atomicAdd(o + 1, v.y * nrm);
    atomicAdd(o + 2, v.z * nrm);
    atomicAdd(o + 3, v.w * nrm);
}

__global__ void k_bias_relu_cvt(const float* __restrict__ acc, const float* __restrict__ b,
                                __bf16* __restrict__ outb, int total) {
    int i = blockIdx.x * blockDim.x + threadIdx.x;
    if (i >= total) return;
    float v = acc[i] + b[i & 127];
    outb[i] = (__bf16)fmaxf(v, 0.0f);
}

// ---------------------------------------------------------------------------
// GAT
// ---------------------------------------------------------------------------
__global__ void k_gat_scores(const float* __restrict__ g, const float* __restrict__ att_src,
                             const float* __restrict__ att_dst, float* __restrict__ a_s,
                             float* __restrict__ a_d, int total /* N*HEADS */) {
    int i = blockIdx.x * blockDim.x + threadIdx.x;
    if (i >= total) return;
    int n = i >> 2, h = i & 3;
    const float* gn = g + (size_t)n * H_DIM + h * HD;
    const float* as = att_src + h * HD;
    const float* ad = att_dst + h * HD;
    float ss = 0.0f, sd = 0.0f;
#pragma unroll 8
    for (int c = 0; c < HD; ++c) {
        float gv = gn[c];
        ss += gv * as[c];
        sd += gv * ad[c];
    }
    a_s[i] = ss;
    a_d[i] = sd;
}

__global__ void k_gat_max_init(const float* __restrict__ a_s, const float* __restrict__ a_d,
                               float* __restrict__ m, int total) {
    int i = blockIdx.x * blockDim.x + threadIdx.x;
    if (i < total) m[i] = leaky02(a_s[i] + a_d[i]);   // self-loop term
}

__global__ void k_gat_edge_max(const int* __restrict__ src, const int* __restrict__ dst,
                               const float* __restrict__ a_s, const float* __restrict__ a_d,
                               float* __restrict__ m, int nE) {
    int i = blockIdx.x * blockDim.x + threadIdx.x;
    int e = i >> 2;
    if (e >= nE) return;
    int h = i & 3;
    int s = src[e], d = dst[e];
    float ev = leaky02(a_s[s * HEADS + h] + a_d[d * HEADS + h]);
    atomicMaxFloat(&m[d * HEADS + h], ev);
}

__global__ void k_gat_self_accum(const float* __restrict__ g, const float* __restrict__ a_s,
                                 const float* __restrict__ a_d, const float* __restrict__ m,
                                 float* __restrict__ num, float* __restrict__ den, int total) {
    int i = blockIdx.x * blockDim.x + threadIdx.x;
    if (i >= total) return;
    int n = i >> 7, c = i & 127, h = c >> 5;
    float es = leaky02(a_s[n * HEADS + h] + a_d[n * HEADS + h]);
    float ex = expf(es - m[n * HEADS + h]);
    num[i] = g[i] * ex;
    if ((c & 31) == 0) den[n * HEADS + h] = ex;
}

__global__ void k_gat_edge_accum(const float* __restrict__ g, const int* __restrict__ src,
                                 const int* __restrict__ dst, const float* __restrict__ a_s,
                                 const float* __restrict__ a_d, const float* __restrict__ m,
                                 float* __restrict__ num, float* __restrict__ den, int nE) {
    int t = blockIdx.x * blockDim.x + threadIdx.x;
    int e = t >> 5;
    if (e >= nE) return;
    int lane = t & 31;
    int head = lane >> 3;             // 4 channels/lane -> 8 lanes/head
    int cbase = lane * 4;
    int s = src[e], d = dst[e];
    float ev = leaky02(a_s[s * HEADS + head] + a_d[d * HEADS + head]);
    float ex = expf(ev - m[d * HEADS + head]);
    float4 gv = *(const float4*)(g + (size_t)s * H_DIM + cbase);
    float* o = num + (size_t)d * H_DIM + cbase;
    atomicAdd(o + 0, gv.x * ex);
    atomicAdd(o + 1, gv.y * ex);
    atomicAdd(o + 2, gv.z * ex);
    atomicAdd(o + 3, gv.w * ex);
    if ((lane & 7) == 0) atomicAdd(&den[d * HEADS + head], ex);
}

__global__ void k_gat_finalize(float* __restrict__ num, const float* __restrict__ den,
                               const float* __restrict__ bg, int total) {
    int i = blockIdx.x * blockDim.x + threadIdx.x;
    if (i >= total) return;
    int n = i >> 7, c = i & 127, h = c >> 5;
    float v = num[i] / den[n * HEADS + h] + bg[c];
    num[i] = v > 0.0f ? v : expf(v) - 1.0f;   // ELU
}

// ---------------------------------------------------------------------------
// Fused MLP head: 128 -> 32 (ReLU) -> 3 (sigmoid). One wave per node.
// ---------------------------------------------------------------------------
__global__ __launch_bounds__(128)
void k_mlp_head(const float* __restrict__ h3, const float* __restrict__ Wf1,
                const float* __restrict__ bf1, const float* __restrict__ Wf2,
                const float* __restrict__ bf2, float* __restrict__ out, int N) {
    int node = blockIdx.x * 4 + (threadIdx.x >> 5);
    if (node >= N) return;
    int lane = threadIdx.x & 31;
    const float* hn = h3 + (size_t)node * H_DIM;
    float acc = bf1[lane];
#pragma unroll 8
    for (int k = 0; k < H_DIM; ++k)
        acc += hn[k] * Wf1[k * 32 + lane];
    float h4 = fmaxf(acc, 0.0f);
#pragma unroll
    for (int o = 0; o < 3; ++o) {
        float p = h4 * Wf2[lane * 3 + o];
        for (int off = 16; off > 0; off >>= 1)
            p += __shfl_down(p, off, 32);
        if (lane == 0)
            out[(size_t)node * 3 + o] = 1.0f / (1.0f + expf(-(p + bf2[o])));
    }
}

// ---------------------------------------------------------------------------
// Host launcher
// ---------------------------------------------------------------------------
extern "C" void kernel_launch(void* const* d_in, const int* in_sizes, int n_in,
                              void* d_out, int out_size, void* d_ws, size_t ws_size,
                              hipStream_t stream) {
    const float* x       = (const float*)d_in[0];
    const int*   eidx    = (const int*)d_in[1];
    const float* W1      = (const float*)d_in[2];
    const float* b1      = (const float*)d_in[3];
    const float* W2      = (const float*)d_in[4];
    const float* b2      = (const float*)d_in[5];
    const float* Wg      = (const float*)d_in[6];
    const float* att_src = (const float*)d_in[7];
    const float* att_dst = (const float*)d_in[8];
    const float* bg      = (const float*)d_in[9];
    const float* Wf1     = (const float*)d_in[10];
    const float* bf1     = (const float*)d_in[11];
    const float* Wf2     = (const float*)d_in[12];
    const float* bf2     = (const float*)d_in[13];

    const int N = in_sizes[0] / F_IN;
    const int E = in_sizes[1] / 2;
    const int* src = eidx;
    const int* dst = eidx + E;
    float* out = (float*)d_out;

    // Workspace carve-out (256B aligned regions)
    size_t off = 0;
    auto wsAlloc = [&](size_t bytes) -> void* {
        void* p = (void*)((char*)d_ws + off);
        off += (bytes + 255) & ~(size_t)255;
        return p;
    };
    float*        dinv = (float*)wsAlloc((size_t)N * 4);
    float*        a_s  = (float*)wsAlloc((size_t)N * HEADS * 4);
    float*        a_d  = (float*)wsAlloc((size_t)N * HEADS * 4);
    float*        mmax = (float*)wsAlloc((size_t)N * HEADS * 4);
    float*        den  = (float*)wsAlloc((size_t)N * HEADS * 4);
    float*        buf0 = (float*)wsAlloc((size_t)N * H_DIM * 4);   // GEMM out / g
    float*        buf1 = (float*)wsAlloc((size_t)N * H_DIM * 4);   // aggregation / h3
    __bf16*       actb = (__bf16*)wsAlloc((size_t)N * H_DIM * 2);  // bf16 activations
    __bf16*       xb   = (__bf16*)wsAlloc((size_t)N * F_IN * 2);
    unsigned int* W1p  = (unsigned int*)wsAlloc((size_t)1 * 8 * 32 * 8 * 4);
    unsigned int* W2p  = (unsigned int*)wsAlloc((size_t)4 * 8 * 32 * 8 * 4);
    unsigned int* Wgp  = (unsigned int*)wsAlloc((size_t)4 * 8 * 32 * 8 * 4);

    const int B = 256;
    const int gN    = (N + B - 1) / B;
    const int gE    = (E + B - 1) / B;
    const int gE4   = (E * 4 + B - 1) / B;
    const int gE32  = (int)(((size_t)E * 32 + B - 1) / B);
    const int gNH   = (N * HEADS + B - 1) / B;
    const int gNC   = (int)(((size_t)N * H_DIM + B - 1) / B);
    const int NC    = N * H_DIM;

    // --- degree / norm -----------------------------------------------------
    k_zero_f32<<<gN, B, 0, stream>>>(dinv, N);
    k_count_deg<<<gE, B, 0, stream>>>(dst, dinv, E);
    k_finalize_dinv<<<gN, B, 0, stream>>>(dinv, N);

    // --- weight packing + input conversion --------------------------------
    k_cvt_f32_bf16<<<(N * F_IN + B - 1) / B, B, 0, stream>>>(x, xb, N * F_IN);
    k_pack_w_bf16<<<8, B, 0, stream>>>(W1, W1p, F_IN, H_DIM);    // 1 chunk * 8 tiles
    k_pack_w_bf16<<<32, B, 0, stream>>>(W2, W2p, H_DIM, H_DIM);  // 4 chunks * 8 tiles
    k_pack_w_bf16<<<32, B, 0, stream>>>(Wg, Wgp, H_DIM, H_DIM);

    dim3 gemmBlk(128);
    dim3 gemmGrd((N / 16 + 3) / 4, H_DIM / 64);

    // --- GCN layer 1 -------------------------------------------------------
    k_wmma_gemm_bf16<<<gemmGrd, gemmBlk, 0, stream>>>(xb, W1p, buf0, N, F_IN, H_DIM);
    k_gcn_self_init<<<gNC, B, 0, stream>>>(buf0, dinv, buf1, NC);
    k_gcn_edge_agg<<<gE32, B, 0, stream>>>(buf0, src, dst, dinv, buf1, E);
    k_bias_relu_cvt<<<gNC, B, 0, stream>>>(buf1, b1, actb, NC);

    // --- GCN layer 2 -------------------------------------------------------
    k_wmma_gemm_bf16<<<gemmGrd, gemmBlk, 0, stream>>>(actb, W2p, buf0, N, H_DIM, H_DIM);
    k_gcn_self_init<<<gNC, B, 0, stream>>>(buf0, dinv, buf1, NC);
    k_gcn_edge_agg<<<gE32, B, 0, stream>>>(buf0, src, dst, dinv, buf1, E);
    k_bias_relu_cvt<<<gNC, B, 0, stream>>>(buf1, b2, actb, NC);

    // --- GAT layer ---------------------------------------------------------
    k_wmma_gemm_bf16<<<gemmGrd, gemmBlk, 0, stream>>>(actb, Wgp, buf0, N, H_DIM, H_DIM); // g
    k_gat_scores<<<gNH, B, 0, stream>>>(buf0, att_src, att_dst, a_s, a_d, N * HEADS);
    k_gat_max_init<<<gNH, B, 0, stream>>>(a_s, a_d, mmax, N * HEADS);
    k_gat_edge_max<<<gE4, B, 0, stream>>>(src, dst, a_s, a_d, mmax, E);
    k_gat_self_accum<<<gNC, B, 0, stream>>>(buf0, a_s, a_d, mmax, buf1, den, NC);
    k_gat_edge_accum<<<gE32, B, 0, stream>>>(buf0, src, dst, a_s, a_d, mmax, buf1, den, E);
    k_gat_finalize<<<gNC, B, 0, stream>>>(buf1, den, bg, NC);    // buf1 = h3 (ELU)

    // --- MLP head ----------------------------------------------------------
    k_mlp_head<<<(N + 3) / 4, 128, 0, stream>>>(buf1, Wf1, bf1, Wf2, bf2, out, N);
}
